// MoE_9483287790085
// MI455X (gfx1250) — compile-verified, matrix operand order checked by reading, not run on hardware
//
#include <hip/hip_runtime.h>

#define D_DIM   1024
#define E_NUM   8
#define BLOCK_N 512   // output columns covered per block (8 waves * 64)
#define WAVE_N  64    // output columns per wave = 4 WMMA subtiles
#define M_TILE  32    // token rows per wave = 2 WMMA row-tiles

typedef __attribute__((ext_vector_type(16))) __bf16 v16bf;
typedef __attribute__((ext_vector_type(8)))  float  v8f;

// ---------------------------------------------------------------------------
// Kernel 1: zero the output accumulator and the per-expert token counters.
// ---------------------------------------------------------------------------
__global__ void moe_zero_kernel(float* __restrict__ out, int n,
                                int* __restrict__ counts) {
  int i = blockIdx.x * blockDim.x + threadIdx.x;
  if (i < n) out[i] = 0.0f;
  if (i < E_NUM) counts[i] = 0;
}

// ---------------------------------------------------------------------------
// Kernel 2: gating. One wave32 per token: 8 partial dot products per lane,
// cross-lane reduce, top-2 over logits (softmax is monotone), then scatter
// the token id into the two selected experts' index lists.
// ---------------------------------------------------------------------------
__global__ void moe_gate_kernel(const float* __restrict__ x,
                                const float* __restrict__ gate_w,
                                const float* __restrict__ gate_b,
                                int* __restrict__ counts,
                                int* __restrict__ lists,
                                int ntok) {
  const int lane = threadIdx.x & 31;
  const int wave = threadIdx.x >> 5;
  const int tok = blockIdx.x * 8 + wave;
  if (tok >= ntok) return;

  float acc[E_NUM];
#pragma unroll
  for (int e = 0; e < E_NUM; ++e) acc[e] = 0.0f;

  const float* xr = x + (size_t)tok * D_DIM;
  for (int d = lane; d < D_DIM; d += 32) {
    float xv = xr[d];
    const float* gr = gate_w + d * E_NUM;   // gate_w is [D][E], 8 contiguous
#pragma unroll
    for (int e = 0; e < E_NUM; ++e) acc[e] += xv * gr[e];
  }

#pragma unroll
  for (int off = 16; off > 0; off >>= 1) {
#pragma unroll
    for (int e = 0; e < E_NUM; ++e)
      acc[e] += __shfl_xor(acc[e], off, 32);
  }

  if (lane == 0) {
#pragma unroll
    for (int e = 0; e < E_NUM; ++e) acc[e] += gate_b[e];
    int b1 = 0, b2 = 0;
    float v1 = acc[0], v2 = -3.4e38f;
#pragma unroll
    for (int e = 1; e < E_NUM; ++e) {
      float v = acc[e];
      if (v > v1)      { v2 = v1; b2 = b1; v1 = v; b1 = e; }
      else if (v > v2) { v2 = v;  b2 = e; }
    }
    int p1 = atomicAdd(&counts[b1], 1);
    lists[b1 * ntok + p1] = tok;
    int p2 = atomicAdd(&counts[b2], 1);
    lists[b2 * ntok + p2] = tok;
  }
}

// ---------------------------------------------------------------------------
// Kernel 3: grouped GEMM over routed tokens. grid = (D/512, tiles32, E).
// 8 waves/block; each wave computes a 32x64 output strip = 2x4 WMMA 16x16
// subtiles. Two A fragments (token rows) are reused across 4 B subtiles,
// and each B fragment feeds 2 WMMAs -> B load+convert per FLOP is halved.
// A/B lane layout per CDNA5 ISA 7.12.2: lanes 0-15 hold K {0..7,16..23},
// lanes 16-31 hold K {8..15,24..31}; A rows / B cols = lane & 15.
// ---------------------------------------------------------------------------
__global__ void __launch_bounds__(256)
moe_expert_gemm_kernel(const float* __restrict__ x,
                       const float* __restrict__ expert_w,
                       const float* __restrict__ expert_b,
                       const int* __restrict__ counts,
                       const int* __restrict__ lists,
                       float* __restrict__ out,
                       int ntok) {
  const int e = blockIdx.z;
  const int cnt = counts[e];
  const int row0 = blockIdx.y * M_TILE;
  if (row0 >= cnt) return;                    // this tile has no tokens

  const int lane  = threadIdx.x & 31;
  const int wave  = threadIdx.x >> 5;
  const int ncol0 = blockIdx.x * BLOCK_N + wave * WAVE_N + (lane & 15);
  const int khalf = lane >> 4;                // which K half this lane owns
  const int mrow  = lane & 15;                // A row within a 16-row tile

  const int* elist = lists + e * ntok;

  // Two token rows per lane (one per 16-row A tile); pad with first row.
  const float* __restrict__ xa[2];
#pragma unroll
  for (int t = 0; t < 2; ++t) {
    int arow = row0 + t * 16 + mrow;
    if (arow >= cnt) arow = row0;
    xa[t] = x + (size_t)elist[arow] * D_DIM;
  }

  const float* __restrict__ wbase = expert_w + (size_t)e * D_DIM * D_DIM;

  v8f c[2][4];
#pragma unroll
  for (int t = 0; t < 2; ++t)
#pragma unroll
    for (int s = 0; s < 4; ++s) c[t][s] = (v8f){};

  for (int kb = 0; kb < D_DIM; kb += 32) {
    if (kb + 32 < D_DIM) {
      __builtin_prefetch(xa[0] + kb + 32, 0, 1);
      __builtin_prefetch(xa[1] + kb + 32, 0, 1);
    }

    // ---- Two A fragments: 16 bf16 K-values of each token row ----
    v16bf a[2];
#pragma unroll
    for (int t = 0; t < 2; ++t) {
      const float4* a0p =
          reinterpret_cast<const float4*>(xa[t] + kb + khalf * 8);
      float4 f0 = a0p[0];
      float4 f1 = a0p[1];
      const float4* a1p =
          reinterpret_cast<const float4*>(xa[t] + kb + 16 + khalf * 8);
      float4 f2 = a1p[0];
      float4 f3 = a1p[1];
      a[t][0]  = (__bf16)f0.x; a[t][1]  = (__bf16)f0.y;
      a[t][2]  = (__bf16)f0.z; a[t][3]  = (__bf16)f0.w;
      a[t][4]  = (__bf16)f1.x; a[t][5]  = (__bf16)f1.y;
      a[t][6]  = (__bf16)f1.z; a[t][7]  = (__bf16)f1.w;
      a[t][8]  = (__bf16)f2.x; a[t][9]  = (__bf16)f2.y;
      a[t][10] = (__bf16)f2.z; a[t][11] = (__bf16)f2.w;
      a[t][12] = (__bf16)f3.x; a[t][13] = (__bf16)f3.y;
      a[t][14] = (__bf16)f3.z; a[t][15] = (__bf16)f3.w;
    }

    // ---- 4 B subtiles; each feeds both A fragments (2 WMMAs) ----
#pragma unroll
    for (int s = 0; s < 4; ++s) {
      const float* __restrict__ wk =
          wbase + (size_t)(kb + khalf * 8) * D_DIM + (ncol0 + s * 16);
      v16bf b;
#pragma unroll
      for (int i = 0; i < 8; ++i) {
        b[i]     = (__bf16)wk[(size_t)i * D_DIM];          // K = kb+khalf*8+i
        b[i + 8] = (__bf16)wk[(size_t)(i + 16) * D_DIM];   // K = +16
      }
      c[0][s] = __builtin_amdgcn_wmma_f32_16x16x32_bf16(
          false, a[0], false, b, (short)0, c[0][s], false, false);
      c[1][s] = __builtin_amdgcn_wmma_f32_16x16x32_bf16(
          false, a[1], false, b, (short)0, c[1][s], false, false);
    }
  }

  // C/D layout: lane holds column (lane&15), rows khalf*8 + r in VGPR r.
  // Hoist the 8 token ids per A tile that this lane-half writes.
  int toks[2][8];
  int valid[2][8];
#pragma unroll
  for (int t = 0; t < 2; ++t) {
#pragma unroll
    for (int r = 0; r < 8; ++r) {
      int gr = row0 + t * 16 + khalf * 8 + r;
      valid[t][r] = (gr < cnt);
      toks[t][r] = elist[valid[t][r] ? gr : row0];
    }
  }

#pragma unroll
  for (int s = 0; s < 4; ++s) {
    const int col = ncol0 + s * 16;
    const float bias = expert_b[e * D_DIM + col];
#pragma unroll
    for (int t = 0; t < 2; ++t) {
#pragma unroll
      for (int r = 0; r < 8; ++r) {
        if (valid[t][r]) {
          atomicAdd(out + (size_t)toks[t][r] * D_DIM + col,
                    c[t][s][r] + bias);
        }
      }
    }
  }
}

// ---------------------------------------------------------------------------
extern "C" void kernel_launch(void* const* d_in, const int* in_sizes, int n_in,
                              void* d_out, int out_size, void* d_ws,
                              size_t ws_size, hipStream_t stream) {
  const float* x        = (const float*)d_in[0];
  const float* gate_w   = (const float*)d_in[1];
  const float* gate_b   = (const float*)d_in[2];
  const float* expert_w = (const float*)d_in[3];
  const float* expert_b = (const float*)d_in[4];
  float* out = (float*)d_out;

  const int ntok = in_sizes[0] / D_DIM;       // B*S = 8192

  // workspace: [0..15] counters (padded), then E per-expert token lists
  int* counts = (int*)d_ws;
  int* lists  = counts + 16;

  // 1) zero output + counters
  {
    int blocks = (out_size + 255) / 256;
    moe_zero_kernel<<<blocks, 256, 0, stream>>>(out, out_size, counts);
  }
  // 2) gating + routing (one wave per token, 8 waves per block)
  {
    int blocks = (ntok + 7) / 8;
    moe_gate_kernel<<<blocks, 256, 0, stream>>>(x, gate_w, gate_b, counts,
                                                lists, ntok);
  }
  // 3) grouped expert GEMM with WMMA bf16 (2x4 subtiles per wave)
  {
    dim3 grid(D_DIM / BLOCK_N, (ntok + M_TILE - 1) / M_TILE, E_NUM);
    moe_expert_gemm_kernel<<<grid, 256, 0, stream>>>(x, expert_w, expert_b,
                                                     counts, lists, out, ntok);
  }
}
